// MultiHeadAttention_10264971837752
// MI455X (gfx1250) — compile-verified, matrix operand order checked by reading, not run on hardware
//
#include <hip/hip_runtime.h>

// ---------------------------------------------------------------------------
// MHA forward for MI455X (gfx1250), wave32 + v_wmma_f32_16x16x32_f16,
// flash-attention stage uses gfx1250 async global->LDS copies (ASYNCcnt).
// Workspace layout (requires ~92.3 MB of d_ws):
//   xh   f16 [B,T,D]          @ 0
//   wt   f16 [3,H,HS,D]       @ 16777216   (Wq,Wk,Wv transposed per head)
//   wpt  f16 [D,D] (Wp^T)     @ 23068672
//   Q    f16 [B,H,T,HS]       @ 25165824   (pre-scaled by 1/sqrt(HS))
//   K    f16 [B,H,T,HS]       @ 41943040
//   Vt   f16 [B,H,HS,T]       @ 58720256
//   Oh   f16 [B,T,H*HS]       @ 75497472
// ---------------------------------------------------------------------------

typedef _Float16 f16;
typedef __attribute__((ext_vector_type(16))) _Float16 v16h;
typedef __attribute__((ext_vector_type(8)))  _Float16 v8h;
typedef __attribute__((ext_vector_type(8)))  float    v8f;
typedef __attribute__((ext_vector_type(4)))  float    v4f;

constexpr int kD  = 1024;
constexpr int kT  = 4096;
constexpr int kB  = 2;
constexpr int kH  = 16;
constexpr int kHS = 64;

// A/B operand: 16 halves per lane = two contiguous runs of 8 halves.
__device__ __forceinline__ v16h ldA16(const f16* p0, const f16* p1) {
  union { v8h h[2]; v16h v; } u;
  u.h[0] = *(const v8h*)p0;
  u.h[1] = *(const v8h*)p1;
  return u.v;
}

__device__ __forceinline__ v8f wmma16(v16h a, v16h b, v8f c) {
  return __builtin_amdgcn_wmma_f32_16x16x32_f16(
      /*neg_a=*/false, a, /*neg_b=*/false, b,
      /*c_mod=*/(short)0, c, /*reuse_a=*/false, /*reuse_b=*/false);
}

// LDS byte-offset of a generic pointer derived from a __shared__ object.
__device__ __forceinline__ unsigned lds_addr_of(const void* p) {
  return (unsigned)(uintptr_t)(__attribute__((address_space(3))) const void*)p;
}

// One per-lane 16B async copy global -> LDS (tracked by ASYNCcnt).
__device__ __forceinline__ void async_b128(unsigned lds, unsigned long long g) {
  asm volatile("global_load_async_to_lds_b128 %0, %1, off"
               :: "v"(lds), "v"(g) : "memory");
}

__device__ __forceinline__ void wait_async_le16() {
  asm volatile("s_wait_asynccnt 16" ::: "memory");
}
__device__ __forceinline__ void wait_async_le0() {
  asm volatile("s_wait_asynccnt 0" ::: "memory");
}

// ---------------------------------------------------------------------------
// Elementwise fp32 -> f16 convert.
__global__ void k_cvt_f16(const float* __restrict__ in, f16* __restrict__ out,
                          long long n) {
  long long i = (long long)blockIdx.x * blockDim.x + threadIdx.x;
  if (i < n) out[i] = (f16)in[i];
}

// Transpose-convert: in [M][R][C] f32  ->  out [M][C][R] f16.
__global__ void k_tr_f16(const float* __restrict__ in, f16* __restrict__ out,
                         int M, int R, int C) {
  long long i = (long long)blockIdx.x * blockDim.x + threadIdx.x;
  long long total = (long long)M * R * C;
  if (i >= total) return;
  int c = (int)(i % C);
  long long t = i / C;
  int r = (int)(t % R);
  int m = (int)(t / R);
  out[((long long)m * C + c) * R + r] = (f16)in[i];
}

// ---------------------------------------------------------------------------
// QKV projection: one wave computes a 16(T) x 64(HS) tile for one head and
// one of {Q,K,V}.  K-loop unrolled x2 with two ping-pong register stages
// (no rotation copies).  grid=(T/16,H,B*3), block=32.
__global__ __launch_bounds__(32)
void k_qkv(const f16* __restrict__ xh, const f16* __restrict__ wt,
           const float* __restrict__ bq, const float* __restrict__ bk,
           const float* __restrict__ bv,
           f16* __restrict__ Q, f16* __restrict__ K, f16* __restrict__ Vt) {
  __shared__ __align__(16) f16 st[16 * 64];

  const int lane = threadIdx.x & 31;
  const int t0   = blockIdx.x * 16;
  const int h    = blockIdx.y;
  const int z    = blockIdx.z;
  const int b    = z / 3;
  const int which = z % 3;                 // 0=Q 1=K 2=V
  const int klo  = (lane & 16) ? 8 : 0;
  const int ln   = lane & 15;
  const int rb   = (lane >> 4) << 3;

  const f16* wsel = wt + ((size_t)which * kH + h) * (size_t)kHS * kD;
  const float* bsel = (which == 0) ? bq : ((which == 1) ? bk : bv);
  const f16* arow = xh + ((size_t)b * kT + t0 + ln) * kD + klo;
  const f16* brow[4];
#pragma unroll
  for (int nt = 0; nt < 4; ++nt)
    brow[nt] = wsel + (size_t)(nt * 16 + ln) * kD + klo;

  v8f acc[4] = {};
  // two-stage pipeline prologue: stage A = k 0, stage B = k 32
  v16h aA = ldA16(arow, arow + 16);
  v16h bA[4];
#pragma unroll
  for (int nt = 0; nt < 4; ++nt) bA[nt] = ldA16(brow[nt], brow[nt] + 16);
  v16h aB = ldA16(arow + 32, arow + 48);
  v16h bB[4];
#pragma unroll
  for (int nt = 0; nt < 4; ++nt) bB[nt] = ldA16(brow[nt] + 32, brow[nt] + 48);

  for (int kk = 64; kk < kD; kk += 64) {
#pragma unroll
    for (int nt = 0; nt < 4; ++nt) acc[nt] = wmma16(aA, bA[nt], acc[nt]);
    aA = ldA16(arow + kk, arow + kk + 16);
#pragma unroll
    for (int nt = 0; nt < 4; ++nt)
      bA[nt] = ldA16(brow[nt] + kk, brow[nt] + kk + 16);
#pragma unroll
    for (int nt = 0; nt < 4; ++nt) acc[nt] = wmma16(aB, bB[nt], acc[nt]);
    aB = ldA16(arow + kk + 32, arow + kk + 48);
#pragma unroll
    for (int nt = 0; nt < 4; ++nt)
      bB[nt] = ldA16(brow[nt] + kk + 32, brow[nt] + kk + 48);
  }
#pragma unroll
  for (int nt = 0; nt < 4; ++nt) acc[nt] = wmma16(aA, bA[nt], acc[nt]);
#pragma unroll
  for (int nt = 0; nt < 4; ++nt) acc[nt] = wmma16(aB, bB[nt], acc[nt]);

  // Bias (per output column), Q additionally scaled by 1/sqrt(HS).
#pragma unroll
  for (int nt = 0; nt < 4; ++nt) {
    float bias = bsel[h * kHS + nt * 16 + ln];
#pragma unroll
    for (int g = 0; g < 8; ++g) {
      acc[nt][g] += bias;
      if (which == 0) acc[nt][g] *= 0.125f;
    }
  }

  const size_t bh = (size_t)b * kH + h;
  if (which == 2) {
    // V stored transposed: Vt[bh][hs][t]; lane owns fixed hs, 8 consecutive t.
#pragma unroll
    for (int nt = 0; nt < 4; ++nt) {
      v8h pk;
#pragma unroll
      for (int g = 0; g < 8; ++g) pk[g] = (f16)acc[nt][g];
      f16* dst = Vt + (bh * kHS + nt * 16 + ln) * (size_t)kT + t0 + rb;
      *(v8h*)dst = pk;
    }
  } else {
    // Stage 16x64 tile through LDS for coalesced row-major stores.
#pragma unroll
    for (int nt = 0; nt < 4; ++nt)
#pragma unroll
      for (int g = 0; g < 8; ++g)
        st[(g + rb) * 64 + nt * 16 + ln] = (f16)acc[nt][g];

    f16* Out = (which == 0) ? Q : K;
    const int m = ln, half = lane >> 4;
    const v8h* src = (const v8h*)&st[m * 64 + half * 32];
    f16* dst = Out + (bh * kT + t0 + m) * (size_t)kHS + half * 32;
#pragma unroll
    for (int q = 0; q < 4; ++q) ((v8h*)dst)[q] = src[q];
  }
}

// ---------------------------------------------------------------------------
// Flash attention: one wave per 16-row query tile; 32 keys per iteration.
// K/V tiles double-buffered in LDS via async global->LDS copies.
// grid = B*H*(T/16)/4 blocks of 128 threads (4 waves).
__global__ __launch_bounds__(128)
void k_attn(const f16* __restrict__ Q, const f16* __restrict__ K,
            const f16* __restrict__ Vt, f16* __restrict__ Oh) {
  __shared__ __align__(16) f16 kbuf[4][2][32 * 64];  // [wave][buf][key][hs]
  __shared__ __align__(16) f16 vbuf[4][2][64 * 32];  // [wave][buf][hs][key]
  __shared__ __align__(16) f16 pst[4][16 * 64];      // P / output staging

  const int wv   = threadIdx.x >> 5;
  const int lane = threadIdx.x & 31;
  const int w    = blockIdx.x * 4 + wv;
  const int i    = w & (kT / 16 - 1);                // query tile 0..255
  const int bh   = w >> 8;                           // 0..B*H-1
  const int klo  = (lane & 16) ? 8 : 0;
  const int ln   = lane & 15;
  const int rb   = (lane >> 4) << 3;
  const float NEGBIG = -1.0e30f;

  const size_t bhT = (size_t)bh * kT;
  const f16* Qrow = Q + (bhT + i * 16 + ln) * (size_t)kHS;
  const unsigned long long Kg =
      (unsigned long long)(uintptr_t)(K + bhT * (size_t)kHS);
  const unsigned long long Vg =
      (unsigned long long)(uintptr_t)(Vt + (size_t)bh * kHS * (size_t)kT);
  f16* ps = &pst[wv][0];

  const unsigned klds[2] = { lds_addr_of(&kbuf[wv][0][0]),
                             lds_addr_of(&kbuf[wv][1][0]) };
  const unsigned vlds[2] = { lds_addr_of(&vbuf[wv][0][0]),
                             lds_addr_of(&vbuf[wv][1][0]) };

  // Issue the 16 async b128 copies that fill one (K,V) tile pair.
  auto issue_tile = [&](int jj, int buf) {
    const int kb = jj * 32;
    // K tile: 32 rows x 128B, contiguous 4KB in global.
    unsigned lk = klds[buf] + lane * 16;
    unsigned long long gk = Kg + (size_t)kb * (kHS * 2) + lane * 16;
#pragma unroll
    for (int ii = 0; ii < 8; ++ii) { async_b128(lk, gk); lk += 512; gk += 512; }
    // V tile: 64 rows (hs) x 64B, global rows strided by T*2 bytes.
    unsigned lv = vlds[buf] + (lane >> 2) * 64 + (lane & 3) * 16;
    unsigned long long gv = Vg + (size_t)(lane >> 2) * (kT * 2)
                               + (size_t)kb * 2 + (lane & 3) * 16;
#pragma unroll
    for (int ii = 0; ii < 8; ++ii) {
      async_b128(lv, gv);
      lv += 512;                       // 8 LDS rows
      gv += (size_t)8 * kT * 2;        // 8 global rows
    }
  };

  // Q tile in A layout, both 32-wide k-chunks of HS=64.
  v16h aq0 = ldA16(Qrow + klo,      Qrow + klo + 16);
  v16h aq1 = ldA16(Qrow + 32 + klo, Qrow + 32 + klo + 16);

  v8f o[4] = {};
  float m_i[8], l_i[8];
#pragma unroll
  for (int g = 0; g < 8; ++g) { m_i[g] = NEGBIG; l_i[g] = 0.f; }

  const int jmax = i >> 1;
  issue_tile(0, 0);                    // prologue

  for (int jj = 0; jj <= jmax; ++jj) {
    const int buf = jj & 1;
    if (jj < jmax) {                   // prefetch next tile, keep 16 in flight
      issue_tile(jj + 1, buf ^ 1);
      wait_async_le16();               // tile jj has landed (in-order)
    } else {
      wait_async_le0();
    }

    // ---- scores: two 16x16 tiles over 32 keys, K-dim = HS = 64 (from LDS)
    const f16* kt = &kbuf[wv][buf][0];
    v8f s0 = {}, s1 = {};
    {
      const f16* k0 = kt + ln * 64 + klo;
      const f16* k1 = kt + (16 + ln) * 64 + klo;
      v16h b0 = ldA16(k0, k0 + 16);
      v16h b1 = ldA16(k1, k1 + 16);
      s0 = wmma16(aq0, b0, s0);
      s1 = wmma16(aq0, b1, s1);
      v16h b2 = ldA16(k0 + 32, k0 + 48);
      v16h b3 = ldA16(k1 + 32, k1 + 48);
      s0 = wmma16(aq1, b2, s0);
      s1 = wmma16(aq1, b3, s1);
    }

    // ---- causal mask (diagonal iteration only; wave-uniform branch)
    if (jj == jmax) {
      const int col0 = jj * 32 + ln;
      const int col1 = col0 + 16;
#pragma unroll
      for (int g = 0; g < 8; ++g) {
        const int rowg = i * 16 + g + rb;
        if (col0 > rowg) s0[g] = NEGBIG;
        if (col1 > rowg) s1[g] = NEGBIG;
      }
    }

    // ---- online softmax update + stage P (f16) into LDS row-major [16x32]
#pragma unroll
    for (int g = 0; g < 8; ++g) {
      float rmax = fmaxf(s0[g], s1[g]);
      rmax = fmaxf(rmax, __shfl_xor(rmax, 1));
      rmax = fmaxf(rmax, __shfl_xor(rmax, 2));
      rmax = fmaxf(rmax, __shfl_xor(rmax, 4));
      rmax = fmaxf(rmax, __shfl_xor(rmax, 8));
      float mnew = fmaxf(m_i[g], rmax);
      float corr = __expf(m_i[g] - mnew);
      float p0 = __expf(s0[g] - mnew);
      float p1 = __expf(s1[g] - mnew);
      float rs = p0 + p1;
      rs += __shfl_xor(rs, 1);
      rs += __shfl_xor(rs, 2);
      rs += __shfl_xor(rs, 4);
      rs += __shfl_xor(rs, 8);
      l_i[g] = l_i[g] * corr + rs;
      m_i[g] = mnew;
      o[0][g] *= corr; o[1][g] *= corr; o[2][g] *= corr; o[3][g] *= corr;
      const int rowg = g + rb;
      ps[rowg * 32 + ln]      = (f16)p0;
      ps[rowg * 32 + 16 + ln] = (f16)p1;
    }

    // ---- P (16x32, A layout from LDS) x V tile (32x64 from LDS)
    const f16* pr = &ps[ln * 32 + klo];
    v16h ap = ldA16(pr, pr + 16);
    const f16* vt_ = &vbuf[wv][buf][0];
#pragma unroll
    for (int nt = 0; nt < 4; ++nt) {
      const f16* vrow = vt_ + (nt * 16 + ln) * 32 + klo;
      v16h bv_ = ldA16(vrow, vrow + 16);
      o[nt] = wmma16(ap, bv_, o[nt]);
    }
  }

  // ---- epilogue: normalize, stage 16x64 f16 tile, coalesced store to Oh
#pragma unroll
  for (int nt = 0; nt < 4; ++nt)
#pragma unroll
    for (int g = 0; g < 8; ++g)
      ps[(g + rb) * 64 + nt * 16 + ln] = (f16)(o[nt][g] / l_i[g]);

  const int b = bh >> 4, h = bh & 15;
  const int m = ln, half = lane >> 4;
  const v8h* src = (const v8h*)&ps[m * 64 + half * 32];
  f16* dst = Oh + ((size_t)b * kT + i * 16 + m) * (size_t)(kH * kHS)
                + h * kHS + half * 32;
#pragma unroll
  for (int q = 0; q < 4; ++q) ((v8h*)dst)[q] = src[q];
}

// ---------------------------------------------------------------------------
// Output projection: Y = Oh * Wp + bp, fp32 out.  K-loop unrolled x2 with
// ping-pong register stages.  grid = (B*T/16, D/64), block = 32.
__global__ __launch_bounds__(32)
void k_proj(const f16* __restrict__ Oh, const f16* __restrict__ Wpt,
            const float* __restrict__ bp, float* __restrict__ out) {
  __shared__ __align__(16) float stf[16 * 64];

  const int lane = threadIdx.x & 31;
  const int rt   = blockIdx.x;     // row tile over B*T
  const int ng   = blockIdx.y;     // 64-col group
  const int klo  = (lane & 16) ? 8 : 0;
  const int ln   = lane & 15;
  const int rb   = (lane >> 4) << 3;

  const f16* arow = Oh + ((size_t)rt * 16 + ln) * kD + klo;
  const f16* brow[4];
#pragma unroll
  for (int nt = 0; nt < 4; ++nt)
    brow[nt] = Wpt + (size_t)(ng * 64 + nt * 16 + ln) * kD + klo;

  v8f acc[4] = {};
  v16h aA = ldA16(arow, arow + 16);
  v16h bA[4];
#pragma unroll
  for (int nt = 0; nt < 4; ++nt) bA[nt] = ldA16(brow[nt], brow[nt] + 16);
  v16h aB = ldA16(arow + 32, arow + 48);
  v16h bB[4];
#pragma unroll
  for (int nt = 0; nt < 4; ++nt) bB[nt] = ldA16(brow[nt] + 32, brow[nt] + 48);

  for (int kk = 64; kk < kD; kk += 64) {
#pragma unroll
    for (int nt = 0; nt < 4; ++nt) acc[nt] = wmma16(aA, bA[nt], acc[nt]);
    aA = ldA16(arow + kk, arow + kk + 16);
#pragma unroll
    for (int nt = 0; nt < 4; ++nt)
      bA[nt] = ldA16(brow[nt] + kk, brow[nt] + kk + 16);
#pragma unroll
    for (int nt = 0; nt < 4; ++nt) acc[nt] = wmma16(aB, bB[nt], acc[nt]);
    aB = ldA16(arow + kk + 32, arow + kk + 48);
#pragma unroll
    for (int nt = 0; nt < 4; ++nt)
      bB[nt] = ldA16(brow[nt] + kk + 32, brow[nt] + kk + 48);
  }
#pragma unroll
  for (int nt = 0; nt < 4; ++nt) acc[nt] = wmma16(aA, bA[nt], acc[nt]);
#pragma unroll
  for (int nt = 0; nt < 4; ++nt) acc[nt] = wmma16(aB, bB[nt], acc[nt]);

#pragma unroll
  for (int nt = 0; nt < 4; ++nt) {
    float bias = bp[ng * 64 + nt * 16 + ln];
#pragma unroll
    for (int g = 0; g < 8; ++g)
      stf[(g + rb) * 64 + nt * 16 + ln] = acc[nt][g] + bias;
  }

  const int m = ln, half = lane >> 4;
  const v4f* src = (const v4f*)&stf[m * 64 + half * 32];
  float* dst = out + ((size_t)rt * 16 + m) * kD + ng * 64 + half * 32;
#pragma unroll
  for (int q = 0; q < 8; ++q) ((v4f*)dst)[q] = src[q];
}

// ---------------------------------------------------------------------------
extern "C" void kernel_launch(void* const* d_in, const int* in_sizes, int n_in,
                              void* d_out, int out_size, void* d_ws,
                              size_t ws_size, hipStream_t stream) {
  (void)in_sizes; (void)n_in; (void)out_size; (void)ws_size;

  const float* x  = (const float*)d_in[0];
  const float* Wq = (const float*)d_in[1];
  const float* Wk = (const float*)d_in[2];
  const float* Wv = (const float*)d_in[3];
  const float* bq = (const float*)d_in[4];
  const float* bk = (const float*)d_in[5];
  const float* bv = (const float*)d_in[6];
  const float* Wp = (const float*)d_in[7];
  const float* bp = (const float*)d_in[8];
  float* out = (float*)d_out;

  char* ws = (char*)d_ws;
  const size_t OFF_XH  = 0;
  const size_t OFF_WT  = 16777216;
  const size_t OFF_WPT = 23068672;
  const size_t OFF_Q   = 25165824;
  const size_t OFF_K   = 41943040;
  const size_t OFF_VT  = 58720256;
  const size_t OFF_OH  = 75497472;

  f16* xh  = (f16*)(ws + OFF_XH);
  f16* wt  = (f16*)(ws + OFF_WT);
  f16* wpt = (f16*)(ws + OFF_WPT);
  f16* Qb  = (f16*)(ws + OFF_Q);
  f16* Kb  = (f16*)(ws + OFF_K);
  f16* Vtb = (f16*)(ws + OFF_VT);
  f16* Oh  = (f16*)(ws + OFF_OH);

  const long long xn = (long long)kB * kT * kD;              // 8388608
  const long long wn = (long long)kH * kD * kHS;             // 1048576
  const size_t WSZ = (size_t)kH * kHS * kD;                  // elements

  k_cvt_f16<<<(unsigned)((xn + 255) / 256), 256, 0, stream>>>(x, xh, xn);
  k_tr_f16<<<(unsigned)((wn + 255) / 256), 256, 0, stream>>>(Wq, wt + 0 * WSZ, kH, kD, kHS);
  k_tr_f16<<<(unsigned)((wn + 255) / 256), 256, 0, stream>>>(Wk, wt + 1 * WSZ, kH, kD, kHS);
  k_tr_f16<<<(unsigned)((wn + 255) / 256), 256, 0, stream>>>(Wv, wt + 2 * WSZ, kH, kD, kHS);
  k_tr_f16<<<(unsigned)(((long long)kD * kD + 255) / 256), 256, 0, stream>>>(Wp, wpt, 1, kD, kD);

  k_qkv<<<dim3(kT / 16, kH, kB * 3), 32, 0, stream>>>(xh, wt, bq, bk, bv, Qb, Kb, Vtb);

  const int nwaves = kB * kH * (kT / 16);                    // 8192
  k_attn<<<nwaves / 4, 128, 0, stream>>>(Qb, Kb, Vtb, Oh);

  k_proj<<<dim3(kB * kT / 16, kD / 64), 32, 0, stream>>>(Oh, wpt, bp, out);
}
